// xLSTMEncoder_63436666962494
// MI455X (gfx1250) — matrix operand order repaired and checked by reading, not compiled, not run on hardware
//
#include <hip/hip_runtime.h>
#include <hip/hip_bf16.h>

// Problem constants
#define B_    64
#define S_    1500
#define E_    128
#define NH_   4
#define DH_   32
#define KCONV 4
#define PUP_  192
#define EPS_  1e-5f
#define BS_   (B_*S_)

// Workspace requirement: ~296 MB
//  H (f32 BSxE) + XN/XC (bf16 BSxE) + G (f32 [bs][h][4][32]) + packed weights

typedef __attribute__((ext_vector_type(16))) __bf16 v16bf;
typedef __attribute__((ext_vector_type(8)))  float  v8f;

union V16BF { v16bf v; uint4 q[2]; unsigned short s[16]; };
union V8F   { v8f   v; float f[8]; };

__device__ __forceinline__ unsigned short f2bf(float f) {
    unsigned u = __builtin_bit_cast(unsigned, f);
    unsigned r = (u + 0x7FFFu + ((u >> 16) & 1u)) >> 16;  // RNE
    return (unsigned short)r;
}
__device__ __forceinline__ float bf2f(unsigned short h) {
    unsigned u = ((unsigned)h) << 16;
    return __builtin_bit_cast(float, u);
}

__device__ __forceinline__ v8f wmma_bf16(v16bf a, v16bf b, v8f c) {
    // v_wmma_f32_16x16x32_bf16 : D = A(16x32) x B(32x16) + C(16x16 f32)
    return __builtin_amdgcn_wmma_f32_16x16x32_bf16(false, a, false, b,
                                                   (short)0, c, false, false);
}

__device__ __forceinline__ float wred(float v) {
#pragma unroll
    for (int m = 16; m > 0; m >>= 1) v += __shfl_xor(v, m, 32);
    return v;
}

// B-matrix (16-bit, 32x16): lane = N (mod 16); lanes 0-15 hold K0..7,16..23,
// lanes 16-31 hold K8..15,24..31
__device__ __forceinline__ int b_kmap(int lane, int e) {
    return ((lane & 16) ? 8 : 0) + (e < 8 ? e : e + 8);
}

// ---------------- utility kernels ----------------

__global__ void zero_kernel(float* p, int n) {
    int i = blockIdx.x * blockDim.x + threadIdx.x;
    if (i < n) p[i] = 0.f;
}

// Pack row-major W[N][K] (f32) into WMMA-B bf16 layout:
// out[((kt*NT+nt)*32+lane)*16+e] = bf16(W[n][k])
__global__ void pack_b_kernel(const float* __restrict__ W,
                              unsigned short* __restrict__ out, int N, int K) {
    int NT = N / 16, KT = K / 32;
    int total = NT * KT * 512;
    for (int i = blockIdx.x * blockDim.x + threadIdx.x; i < total;
         i += gridDim.x * blockDim.x) {
        int e = i & 15, lane = (i >> 4) & 31, tile = i >> 9;
        int nt = tile % NT, kt = tile / NT;
        int n = nt * 16 + (lane & 15);
        int k = kt * 32 + b_kmap(lane, e);
        out[i] = f2bf(W[(size_t)n * K + k]);
    }
}

__global__ void pack_convw_kernel(const float* __restrict__ w,
                                  float* __restrict__ wT) {
    int i = blockIdx.x * blockDim.x + threadIdx.x;  // E*K
    if (i < E_ * KCONV) {
        int e = i / KCONV, k = i % KCONV;
        wT[k * E_ + e] = w[i];
    }
}

// ---------------- stage 1: embedding + LN1 ----------------
__global__ __launch_bounds__(256) void embed_ln_kernel(
    const float* __restrict__ emb, const int* __restrict__ seq,
    const float* __restrict__ ln1w, float* __restrict__ H,
    unsigned short* __restrict__ XN) {
    int wid = blockIdx.x * 8 + (threadIdx.x >> 5);  // row (b*S+s)
    int lane = threadIdx.x & 31;
    if (wid >= BS_) return;
    int tok = seq[wid];
    float4 x = ((const float4*)(emb + (size_t)tok * E_))[lane];
    float mu = wred(x.x + x.y + x.z + x.w) * (1.f / E_);
    float4 d = {x.x - mu, x.y - mu, x.z - mu, x.w - mu};
    float var = wred(d.x * d.x + d.y * d.y + d.z * d.z + d.w * d.w) * (1.f / E_);
    float rs = rsqrtf(var + EPS_);
    float4 w = ((const float4*)ln1w)[lane];
    ((float4*)(H + (size_t)wid * E_))[lane] = x;
    ushort4 o;
    o.x = f2bf(d.x * rs * w.x);
    o.y = f2bf(d.y * rs * w.y);
    o.z = f2bf(d.z * rs * w.z);
    o.w = f2bf(d.w * rs * w.w);
    ((ushort4*)(XN + (size_t)wid * E_))[lane] = o;
}

// ---------------- stage 2: causal depthwise conv + SiLU ----------------
__global__ __launch_bounds__(256) void conv_silu_kernel(
    const unsigned short* __restrict__ XN, const float* __restrict__ convWT,
    const float* __restrict__ convB, unsigned short* __restrict__ XC) {
    int i = blockIdx.x * blockDim.x + threadIdx.x;  // BS_*32
    if (i >= BS_ * 32) return;
    int e4 = i & 31;
    int bs = i >> 5;
    int t = bs % S_;
    float4 acc = ((const float4*)convB)[e4];
#pragma unroll
    for (int k = 0; k < KCONV; k++) {
        int tt = t - 3 + k;
        if (tt >= 0) {
            ushort4 xr = ((const ushort4*)(XN + (size_t)(bs - 3 + k) * E_))[e4];
            float4 w = ((const float4*)(convWT + k * E_))[e4];
            acc.x += bf2f(xr.x) * w.x;
            acc.y += bf2f(xr.y) * w.y;
            acc.z += bf2f(xr.z) * w.z;
            acc.w += bf2f(xr.w) * w.w;
        }
    }
    acc.x = acc.x / (1.f + __expf(-acc.x));
    acc.y = acc.y / (1.f + __expf(-acc.y));
    acc.z = acc.z / (1.f + __expf(-acc.z));
    acc.w = acc.w / (1.f + __expf(-acc.w));
    ushort4 o = {f2bf(acc.x), f2bf(acc.y), f2bf(acc.z), f2bf(acc.w)};
    ((ushort4*)(XC + (size_t)bs * E_))[e4] = o;
}

// ---------------- stage 3: gate pre-activation GEMMs (WMMA) ----------------
// wave per (head, 64-row unit): B tiles preloaded once, 4 M-subtiles,
// 32x v_wmma_f32_16x16x32_bf16 per wave.
// G layout: [bs][h][4][32] f32 -> linear store addressing (no div/mod)
__global__ __launch_bounds__(256) void gate_gemm_kernel(
    const unsigned short* __restrict__ XC, const unsigned short* __restrict__ XN,
    const unsigned short* __restrict__ WP, float* __restrict__ G) {
    int wid = blockIdx.x * 8 + (threadIdx.x >> 5);
    int lane = threadIdx.x & 31;
    int h = wid & 3;
    int mu4 = wid >> 2;  // 64-row unit index, 0..BS_/64-1
    if (mu4 >= BS_ / 64) return;
    int bs0 = mu4 * 64;

    // preload the 8 B tiles for this head (4 gates x 2 n-tiles)
    V16BF bw[4][2];
    const unsigned short* wbase = WP + (size_t)h * 4 * 1024;
#pragma unroll
    for (int g = 0; g < 4; g++)
#pragma unroll
        for (int nt = 0; nt < 2; nt++) {
            const unsigned short* wp = wbase + g * 1024 + (nt * 32 + lane) * 16;
            bw[g][nt].q[0] = *(const uint4*)(wp);
            bw[g][nt].q[1] = *(const uint4*)(wp + 8);
        }

    int kb = (lane & 16) ? 8 : 0;
    int mhi = (lane >> 4) << 3;  // 0 or 8
#pragma unroll
    for (int sub = 0; sub < 4; sub++) {
        int rb = bs0 + sub * 16;
        size_t rbase = (size_t)(rb + (lane & 15)) * E_ + h * DH_;
        V16BF axc, axn;
        axc.q[0] = *(const uint4*)(XC + rbase + kb);
        axc.q[1] = *(const uint4*)(XC + rbase + 16 + kb);
        axn.q[0] = *(const uint4*)(XN + rbase + kb);
        axn.q[1] = *(const uint4*)(XN + rbase + 16 + kb);
#pragma unroll
        for (int g = 0; g < 4; g++) {
            V16BF a = (g < 2) ? axc : axn;  // i,f from conv path; z,o from LN
#pragma unroll
            for (int nt = 0; nt < 2; nt++) {
                V8F acc;
#pragma unroll
                for (int q = 0; q < 8; q++) acc.f[q] = 0.f;
                acc.v = wmma_bf16(a.v, bw[g][nt].v, acc.v);
                // store: G[((rb+mm)*4 + h)*128 + g*32 + nt*16 + n]
                float* gp = G + (((size_t)rb * 4 + h) * 128 + g * 32 + nt * 16 +
                                 (lane & 15)) + (size_t)mhi * 512;
#pragma unroll
                for (int v = 0; v < 8; v++) gp[(size_t)v * 512] = acc.f[v];
            }
        }
    }
}

// ---------------- stage 4: sequential mLSTM scan ----------------
// one wave per (b,h); R held in 128 VGPRs; y broadcast via readlane;
// fused head-norm + residual add into H
__device__ __forceinline__ float logsigf(float x) {
    return fminf(x, 0.f) - log1pf(__expf(-fabsf(x)));
}

__global__ __launch_bounds__(32) void scan_kernel(
    const float* __restrict__ G, const float* __restrict__ R,
    const float* __restrict__ bcell, const float* __restrict__ gnw,
    float* __restrict__ H) {
    int u = blockIdx.x;  // 0..255
    int b = u >> 2, h = u & 3;
    int j = threadIdx.x;
    float Rr[4][32];
#pragma unroll
    for (int d = 0; d < DH_; d++)
#pragma unroll
        for (int g = 0; g < 4; g++)
            Rr[g][d] = R[(size_t)(h * DH_ + d) * 128 + g * 32 + j];
    float bc[4];
#pragma unroll
    for (int g = 0; g < 4; g++) bc[g] = bcell[(h * 4 + g) * 32 + j];
    float gw = gnw[h * DH_ + j];
    // G layout [bs][h][4][32]: per-step stride 512 floats
    const float* Gp = G + ((size_t)(b * S_) * 4 + h) * 128;
    float* Hp = H + (size_t)b * S_ * E_ + h * DH_ + j;
    float y = 0.f, c = 0.f, n = 0.f, mst = 0.f;
    for (int t = 0; t < S_; t++) {
        const float* gp = Gp + (size_t)t * 512;
        float gi = gp[j], gf = gp[32 + j], gz = gp[64 + j], go = gp[96 + j];
        float ri = 0.f, rf = 0.f, rz = 0.f, ro = 0.f;
#pragma unroll
        for (int d = 0; d < DH_; d++) {
            float yd = __shfl(y, d, 32);  // v_readlane broadcast
            ri = fmaf(yd, Rr[0][d], ri);
            rf = fmaf(yd, Rr[1][d], rf);
            rz = fmaf(yd, Rr[2][d], rz);
            ro = fmaf(yd, Rr[3][d], ro);
        }
        float iraw = gi + ri + bc[0];
        float fraw = gf + rf + bc[1];
        float zraw = gz + rz + bc[2];
        float oraw = go + ro + bc[3];
        float lf = mst + logsigf(fraw);
        float mnew = fmaxf(iraw, lf);
        float ig = __expf(iraw - mnew);
        float fg = __expf(lf - mnew);
        c = fg * c + ig * tanhf(zraw);
        n = fg * n + ig;
        y = (c / n) / (1.f + __expf(-oraw));
        mst = mnew;
        // head-norm over the 32 lanes of this head
        float mu = wred(y) * (1.f / DH_);
        float dv = y - mu;
        float var = wred(dv * dv) * (1.f / DH_);
        float hn = dv * rsqrtf(var + EPS_) * gw;
        float* hp = Hp + (size_t)t * E_;
        *hp = *hp + hn;  // unique (b,t,h,d) per lane -> race-free
    }
}

// ---------------- stage 5: LN2 + FFN(WMMA) + residual + LNpost + mean ----------
#define NT_UP 24
#define KT_UP 4
#define NT_DN 8
#define KT_DN 6

__global__ __launch_bounds__(256) void ffn_kernel(
    const float* __restrict__ H, const float* __restrict__ ln2w,
    const float* __restrict__ lnpw, const unsigned short* __restrict__ UPW,
    const unsigned short* __restrict__ DNW, float* __restrict__ OUT) {
    __shared__ __align__(32) float hrow[16 * 128];
    __shared__ __align__(32) unsigned short a_up[KT_UP * 32 * 16];
    __shared__ __align__(32) float up_ls[16 * 384];
    __shared__ __align__(32) unsigned short a_dn[KT_DN * 32 * 16];
    __shared__ __align__(32) float hout[16 * 128];
    int b = blockIdx.x, st = blockIdx.y;
    int s0 = st * 16;
    int tid = threadIdx.x;
    int wv = tid >> 5, lane = tid & 31;

    {  // load 16 rows of h (zero-pad past S)
        int r = tid >> 4, c0 = (tid & 15) * 8;
        int s = s0 + r;
        float4 v0 = {0, 0, 0, 0}, v1 = {0, 0, 0, 0};
        if (s < S_) {
            const float4* p = (const float4*)(H + ((size_t)b * S_ + s) * E_ + c0);
            v0 = p[0];
            v1 = p[1];
        }
        *(float4*)(hrow + r * 128 + c0) = v0;
        *(float4*)(hrow + r * 128 + c0 + 4) = v1;
    }
    __syncthreads();

    // LN2 -> bf16 A-layout in LDS
#pragma unroll
    for (int rr = 0; rr < 2; rr++) {
        int r = wv * 2 + rr;
        int c0 = lane * 4;
        float4 x = *(const float4*)(hrow + r * 128 + c0);
        float mu = wred(x.x + x.y + x.z + x.w) * (1.f / E_);
        float4 d = {x.x - mu, x.y - mu, x.z - mu, x.w - mu};
        float var = wred(d.x * d.x + d.y * d.y + d.z * d.z + d.w * d.w) * (1.f / E_);
        float rs = rsqrtf(var + EPS_);
        float4 w = ((const float4*)ln2w)[lane];
        float vals[4] = {d.x * rs * w.x, d.y * rs * w.y, d.z * rs * w.z,
                         d.w * rs * w.w};
#pragma unroll
        for (int q = 0; q < 4; q++) {
            int cc = c0 + q;
            int kt = cc >> 5, kp = cc & 31;
            int l = r + ((kp & 8) ? 16 : 0);
            int e = (kp & 7) + ((kp & 16) ? 8 : 0);
            a_up[(kt * 32 + l) * 16 + e] = f2bf(vals[q]);
        }
    }
    __syncthreads();

    // up GEMM: wave wv computes n-tiles 3wv..3wv+2 (12 WMMAs/wave)
    V8F acc[3];
#pragma unroll
    for (int i = 0; i < 3; i++)
#pragma unroll
        for (int q = 0; q < 8; q++) acc[i].f[q] = 0.f;
#pragma unroll
    for (int kt = 0; kt < KT_UP; kt++) {
        V16BF a;
        a.q[0] = *(const uint4*)(a_up + (kt * 32 + lane) * 16);
        a.q[1] = *(const uint4*)(a_up + (kt * 32 + lane) * 16 + 8);
#pragma unroll
        for (int i = 0; i < 3; i++) {
            int nt = wv * 3 + i;
            V16BF bm;
            const unsigned short* wp =
                UPW + ((size_t)(kt * NT_UP + nt) * 32 + lane) * 16;
            bm.q[0] = *(const uint4*)(wp);
            bm.q[1] = *(const uint4*)(wp + 8);
            acc[i].v = wmma_bf16(a.v, bm.v, acc[i].v);
        }
    }
#pragma unroll
    for (int i = 0; i < 3; i++)
#pragma unroll
        for (int v = 0; v < 8; v++) {
            int m = v + ((lane >> 4) << 3);
            up_ls[m * 384 + (wv * 3 + i) * 16 + (lane & 15)] = acc[i].f[v];
        }
    __syncthreads();

    // exact GeLU gate * up, pack to A-layout for down GEMM
    for (int idx = tid; idx < 16 * PUP_; idx += 256) {
        int r = idx / PUP_, cc = idx % PUP_;
        float g = up_ls[r * 384 + cc];
        float uu = up_ls[r * 384 + PUP_ + cc];
        float act = 0.5f * g * (1.f + erff(g * 0.70710678118f)) * uu;
        int kt = cc >> 5, kp = cc & 31;
        int l = r + ((kp & 8) ? 16 : 0);
        int e = (kp & 7) + ((kp & 16) ? 8 : 0);
        a_dn[(kt * 32 + l) * 16 + e] = f2bf(act);
    }
    __syncthreads();

    // down GEMM: wave wv computes n-tile wv (6 WMMAs/wave) + residual
    V8F dacc;
#pragma unroll
    for (int q = 0; q < 8; q++) dacc.f[q] = 0.f;
#pragma unroll
    for (int kt = 0; kt < KT_DN; kt++) {
        V16BF a;
        a.q[0] = *(const uint4*)(a_dn + (kt * 32 + lane) * 16);
        a.q[1] = *(const uint4*)(a_dn + (kt * 32 + lane) * 16 + 8);
        V16BF bm;
        const unsigned short* wp =
            DNW + ((size_t)(kt * NT_DN + wv) * 32 + lane) * 16;
        bm.q[0] = *(const uint4*)(wp);
        bm.q[1] = *(const uint4*)(wp + 8);
        dacc.v = wmma_bf16(a.v, bm.v, dacc.v);
    }
#pragma unroll
    for (int v = 0; v < 8; v++) {
        int m = v + ((lane >> 4) << 3);
        int col = wv * 16 + (lane & 15);
        hout[m * 128 + col] = dacc.f[v] + hrow[m * 128 + col];
    }
    __syncthreads();

    // LN post (in-place per-row)
#pragma unroll
    for (int rr = 0; rr < 2; rr++) {
        int r = wv * 2 + rr;
        int c0 = lane * 4;
        float4 x = *(const float4*)(hout + r * 128 + c0);
        float mu = wred(x.x + x.y + x.z + x.w) * (1.f / E_);
        float4 d = {x.x - mu, x.y - mu, x.z - mu, x.w - mu};
        float var = wred(d.x * d.x + d.y * d.y + d.z * d.z + d.w * d.w) * (1.f / E_);
        float rs = rsqrtf(var + EPS_);
        float4 w = ((const float4*)lnpw)[lane];
        float4 o = {d.x * rs * w.x, d.y * rs * w.y, d.z * rs * w.z, d.w * rs * w.w};
        *(float4*)(hout + r * 128 + c0) = o;
    }
    __syncthreads();

    // per-column partial mean -> global atomic accumulate
    if (tid < 128) {
        float sum = 0.f;
#pragma unroll
        for (int r = 0; r < 16; r++)
            if (s0 + r < S_) sum += hout[r * 128 + tid];
        atomicAdd(OUT + b * E_ + tid, sum * (1.f / S_));
    }
}

// ---------------- launcher ----------------
extern "C" void kernel_launch(void* const* d_in, const int* in_sizes, int n_in,
                              void* d_out, int out_size, void* d_ws,
                              size_t ws_size, hipStream_t stream) {
    (void)in_sizes; (void)n_in; (void)out_size; (void)ws_size;
    const float* emb   = (const float*)d_in[0];
    const float* convw = (const float*)d_in[1];
    const float* convb = (const float*)d_in[2];
    const float* Wi    = (const float*)d_in[3];
    const float* Wf    = (const float*)d_in[4];
    const float* Wz    = (const float*)d_in[5];
    const float* Wo    = (const float*)d_in[6];
    const float* R     = (const float*)d_in[7];
    const float* bcell = (const float*)d_in[8];
    const float* gnw   = (const float*)d_in[9];
    const float* ln1w  = (const float*)d_in[10];
    const float* ln2w  = (const float*)d_in[11];
    const float* lnpw  = (const float*)d_in[12];
    const float* upw   = (const float*)d_in[13];
    const float* dnw   = (const float*)d_in[14];
    const int*   seq   = (const int*)d_in[15];
    float* out = (float*)d_out;

    char* ws = (char*)d_ws;
    size_t off = 0;
    auto take = [&](size_t bytes) -> char* {
        char* p = ws + off;
        off = (off + bytes + 255) & ~(size_t)255;
        return p;
    };
    float*          H   = (float*)take((size_t)BS_ * E_ * 4);
    unsigned short* XN  = (unsigned short*)take((size_t)BS_ * E_ * 2);
    unsigned short* XC  = (unsigned short*)take((size_t)BS_ * E_ * 2);
    float*          G   = (float*)take((size_t)BS_ * NH_ * 4 * DH_ * 4);
    unsigned short* GW  = (unsigned short*)take(16 * 1024 * 2);
    unsigned short* UPW = (unsigned short*)take((size_t)KT_UP * NT_UP * 512 * 2);
    unsigned short* DNW = (unsigned short*)take((size_t)KT_DN * NT_DN * 512 * 2);
    float*          CWT = (float*)take(KCONV * E_ * 4);

    zero_kernel<<<(B_ * E_ + 255) / 256, 256, 0, stream>>>(out, B_ * E_);
    pack_convw_kernel<<<2, 256, 0, stream>>>(convw, CWT);
    const float* Wg[4] = {Wi, Wf, Wz, Wo};
    for (int h = 0; h < NH_; h++)
        for (int g = 0; g < 4; g++)
            pack_b_kernel<<<4, 256, 0, stream>>>(Wg[g] + h * DH_ * DH_,
                                                 GW + (h * 4 + g) * 1024, DH_, DH_);
    pack_b_kernel<<<192, 256, 0, stream>>>(upw, UPW, 2 * PUP_, E_);
    pack_b_kernel<<<96, 256, 0, stream>>>(dnw, DNW, E_, PUP_);

    embed_ln_kernel<<<BS_ / 8, 256, 0, stream>>>(emb, seq, ln1w, H, XN);
    conv_silu_kernel<<<(BS_ * 32 + 255) / 256, 256, 0, stream>>>(XN, CWT, convb, XC);
    // 4 heads x (BS/64) 64-row units, 8 waves per block
    gate_gemm_kernel<<<(BS_ / 64) * 4 / 8, 256, 0, stream>>>(XC, XN, GW, G);
    scan_kernel<<<B_ * NH_, 32, 0, stream>>>(G, R, bcell, gnw, H);
    ffn_kernel<<<dim3(B_, (S_ + 15) / 16), 256, 0, stream>>>(H, ln2w, lnpw, UPW,
                                                             DNW, out);
}
